// AttentionHead_86672440033524
// MI455X (gfx1250) — matrix operand order repaired and checked by reading, not compiled, not run on hardware
//
#include <hip/hip_runtime.h>
#include <math.h>

// ---------------------------------------------------------------------------
// Fused attention head for MI455X (gfx1250, wave32, WMMA).
//   pass 0: Wq/Wk/Wv -> f16 copies (converted once, L2-resident)
//   pass 1: Q = (q Wq^T + bq)/sqrt(128) -> f16 [B*S,128]
//           K =  k Wk^T + bk            -> f16 [B*S,128]
//           V =  v Wv^T + bv            -> f16 TRANSPOSED [B,128,S]
//   pass 2: flash attention with causal + global-token mask, f32 accum.
// Workspace: 3*16384*128*2 (QKV f16) + 3*128*1024*2 (W f16) = 13.37 MB.
// ---------------------------------------------------------------------------

typedef __attribute__((ext_vector_type(16))) _Float16 v16h;
typedef __attribute__((ext_vector_type(8)))  _Float16 v8h;
typedef __attribute__((ext_vector_type(8)))  float    v8f;

#define SEQ   2048
#define BATCH 8
#define DIN   1024
#define DQK   128

// Global-token indices: must equal np.random.default_rng(0).choice(2048, 32, False).
// (Placeholder values — fixed constant table, swap in the exact RNG output.)
__constant__ int GTOK[32] = {
    85, 1217,  182,  460, 1874,  993, 1729, 1023,
    57, 1308,  641,  222, 1599,  734,  412, 2011,
     5,  888, 1456,  367, 1040, 1777,  301, 1688,
   920,  133,  547, 1950,  812, 1135,  664, 1501
};

__device__ __forceinline__ v8f wmma_f16(v16h a, v16h b, v8f c) {
  // D = A(16x32 f16) * B(32x16 f16) + C(16x16 f32)
  return __builtin_amdgcn_wmma_f32_16x16x32_f16(
      /*neg_a=*/false, a, /*neg_b=*/false, b,
      /*c_mod=*/(short)0, c, /*reuse_a=*/false, /*reuse_b=*/false);
}

// ---------------------------------------------------------------------------
// One-shot fp32 -> f16 conversion (for the three weight matrices).
// Each thread converts 8 contiguous elements: 2x float4 load, 1x v8h store.
// ---------------------------------------------------------------------------
__global__ __launch_bounds__(256) void cvt_f16_kernel(
    const float* __restrict__ src, _Float16* __restrict__ dst)
{
  const size_t i = ((size_t)blockIdx.x * blockDim.x + threadIdx.x) * 8;
  float4 a = *(const float4*)(src + i);
  float4 b = *(const float4*)(src + i + 4);
  v8h h;
  h[0] = (_Float16)a.x; h[1] = (_Float16)a.y; h[2] = (_Float16)a.z; h[3] = (_Float16)a.w;
  h[4] = (_Float16)b.x; h[5] = (_Float16)b.y; h[6] = (_Float16)b.z; h[7] = (_Float16)b.w;
  *(v8h*)(dst + i) = h;
}

// ---------------------------------------------------------------------------
// Projection GEMM: out[m][n] = sum_k X[m][k] * Wh[n][k] + bias[n], scaled,
// stored as f16 either row-major [m][128] or transposed [b][n][s].
// Block: 256 threads = 8 waves; each wave: 16 rows x 128 cols, K steps of 32.
// Weights arrive pre-converted to f16 -> B fragment is one aligned 32B load.
// ---------------------------------------------------------------------------
__global__ __launch_bounds__(256) void proj_kernel(
    const float* __restrict__ X, const _Float16* __restrict__ Wh,
    const float* __restrict__ bias, _Float16* __restrict__ out,
    int transposed, float scale)
{
  const int tid  = threadIdx.x;
  const int wave = tid >> 5;
  const int lane = tid & 31;
  const int r    = lane & 15;   // row (A) / col (B,C) index within 16
  const int hg   = lane >> 4;   // half-wave group selects K sub-range
  const int row0 = blockIdx.x * 128 + wave * 16;

  v8f acc[8] = {};
  const float* xrow = X + (size_t)(row0 + r) * DIN;

  for (int kb = 0; kb < DIN; kb += 32) {
    // A fragment: lane groups hold K = kb+hg*16+{0..15} (contiguous)
    v16h a;
    const float* p0 = xrow + kb + hg * 16;
    const float* p1 = xrow + kb + hg * 16 + 8;
#pragma unroll
    for (int i = 0; i < 8; ++i) { a[i] = (_Float16)p0[i]; a[8 + i] = (_Float16)p1[i]; }

#pragma unroll
    for (int nt = 0; nt < 8; ++nt) {
      // B fragment: B[k][n] = Wh[n][k]; lane = col n, 16 contiguous k (32B).
      v16h bf = *(const v16h*)(Wh + (size_t)(nt * 16 + r) * DIN + kb + hg * 16);
      acc[nt] = wmma_f16(a, bf, acc[nt]);
    }
  }

#pragma unroll
  for (int nt = 0; nt < 8; ++nt) {
    const int col = nt * 16 + r;
    const float bv = bias[col];
#pragma unroll
    for (int j = 0; j < 8; ++j) {               // C layout: row = j + 8*hg
      const int row = row0 + j + 8 * hg;
      const _Float16 h = (_Float16)((acc[nt][j] + bv) * scale);
      if (!transposed) {
        out[(size_t)row * DQK + col] = h;
      } else {
        const int bb = row >> 11, s = row & (SEQ - 1);
        out[((size_t)bb * DQK + col) * SEQ + s] = h;
      }
    }
  }
}

// ---------------------------------------------------------------------------
// Flash attention: block = 128 query rows (8 waves x 16), key tiles of 32.
// ---------------------------------------------------------------------------
__global__ __launch_bounds__(256) void attn_kernel(
    const _Float16* __restrict__ Qh, const _Float16* __restrict__ Kh,
    const _Float16* __restrict__ Vt, float* __restrict__ Out)
{
  __shared__ uint32_t gmask[SEQ / 32];                     // global-token bitmap
  __shared__ __align__(32) _Float16 pst_all[8][16 * 32];   // per-wave P staging

  const int tid = threadIdx.x;
  if (tid < SEQ / 32) {
    uint32_t w = 0;
#pragma unroll
    for (int i = 0; i < 32; ++i) {
      const int g = GTOK[i];
      w |= ((g >> 5) == tid) ? (1u << (g & 31)) : 0u;
    }
    gmask[tid] = w;
  }
  __syncthreads();

  const int wave = tid >> 5, lane = tid & 31, r = lane & 15, hg = lane >> 4;
  const int b     = blockIdx.x >> 4;
  const int qt    = blockIdx.x & 15;
  const int qbase = qt * 128;
  const int qrow0 = qbase + wave * 16;
  _Float16* pst = &pst_all[wave][0];

  const _Float16* Qb = Qh + (size_t)b * SEQ * DQK;
  const _Float16* Kb = Kh + (size_t)b * SEQ * DQK;
  const _Float16* Vb = Vt + (size_t)b * DQK * SEQ;

  // Preload Q fragments (16 rows x 128 d) as 4 K=32 A-fragments.
  v16h qa[4];
  {
    const _Float16* qrow = Qb + (size_t)(qrow0 + r) * DQK;
#pragma unroll
    for (int c = 0; c < 4; ++c) {
      v8h lo = *(const v8h*)(qrow + c * 32 + hg * 16);
      v8h hi = *(const v8h*)(qrow + c * 32 + hg * 16 + 8);
      v16h a;
#pragma unroll
      for (int i = 0; i < 8; ++i) { a[i] = lo[i]; a[8 + i] = hi[i]; }
      qa[c] = a;
    }
  }

  int  rowq[8];
  bool qglob[8];
#pragma unroll
  for (int j = 0; j < 8; ++j) {
    rowq[j]  = qrow0 + j + 8 * hg;
    qglob[j] = (gmask[rowq[j] >> 5] >> (rowq[j] & 31)) & 1u;
  }

  // If any query row in this block is a global token it attends to all keys.
  bool anyg = false;
#pragma unroll
  for (int w = 0; w < 4; ++w) anyg = anyg || (gmask[(qbase >> 5) + w] != 0u);
  const int nkt = anyg ? (SEQ / 32) : (qt * 4 + 4);   // causal bound otherwise

  float m_i[8], l_i[8];
#pragma unroll
  for (int j = 0; j < 8; ++j) { m_i[j] = -INFINITY; l_i[j] = 0.0f; }
  v8f oacc[8] = {};

  for (int kt = 0; kt < nkt; ++kt) {
    const int k0 = kt * 32;
    if (kt + 1 < nkt) {
      __builtin_prefetch(Kb + (size_t)(k0 + 32) * DQK, 0, 1);
      __builtin_prefetch(Vb + (size_t)r * SEQ + k0 + 32, 0, 1);
    }

    // ---- scores S = (Q/sqrt(d)) K^T for 32 keys (two 16-wide C tiles) ----
    v8f s0 = {}, s1 = {};
    {
      const _Float16* kr0 = Kb + (size_t)(k0 + r) * DQK;
      const _Float16* kr1 = Kb + (size_t)(k0 + 16 + r) * DQK;
#pragma unroll
      for (int c = 0; c < 4; ++c) {
        v16h b0 = *(const v16h*)(kr0 + c * 32 + hg * 16);
        s0 = wmma_f16(qa[c], b0, s0);
      }
#pragma unroll
      for (int c = 0; c < 4; ++c) {
        v16h b1 = *(const v16h*)(kr1 + c * 32 + hg * 16);
        s1 = wmma_f16(qa[c], b1, s1);
      }
    }

    // ---- mask: allowed if key<=query, key global, or query global ----
    const int  key0 = k0 + r, key1 = k0 + 16 + r;
    const bool kg0 = (gmask[key0 >> 5] >> (key0 & 31)) & 1u;
    const bool kg1 = (gmask[key1 >> 5] >> (key1 & 31)) & 1u;
#pragma unroll
    for (int j = 0; j < 8; ++j) {
      if (!(kg0 | qglob[j] | (key0 <= rowq[j]))) s0[j] = -INFINITY;
      if (!(kg1 | qglob[j] | (key1 <= rowq[j]))) s1[j] = -INFINITY;
    }

    // ---- online softmax (row reductions across the 16-lane half) ----
    float corr[8];
#pragma unroll
    for (int j = 0; j < 8; ++j) {
      float mx = fmaxf(s0[j], s1[j]);
#pragma unroll
      for (int off = 8; off >= 1; off >>= 1) mx = fmaxf(mx, __shfl_xor(mx, off, 16));
      const float mnew = fmaxf(m_i[j], mx);
      const float p0 = __expf(s0[j] - mnew);
      const float p1 = __expf(s1[j] - mnew);
      corr[j] = __expf(m_i[j] - mnew);
      m_i[j]  = mnew;
      float rs = p0 + p1;
#pragma unroll
      for (int off = 8; off >= 1; off >>= 1) rs += __shfl_xor(rs, off, 16);
      l_i[j] = l_i[j] * corr[j] + rs;
      const int rl = j + 8 * hg;              // C-layout row -> LDS [row][key]
      pst[rl * 32 + r]      = (_Float16)p0;
      pst[rl * 32 + 16 + r] = (_Float16)p1;
    }
#pragma unroll
    for (int nt = 0; nt < 8; ++nt)
#pragma unroll
      for (int j = 0; j < 8; ++j) oacc[nt][j] *= corr[j];

    // same-wave LDS RAW: drain DS counter before re-reading staged P
    asm volatile("s_wait_dscnt 0" ::: "memory");

    // ---- P A-fragment (16x32) from LDS (C->A layout transpose) ----
    v16h pa;
    {
      v8h plo = *(const v8h*)(pst + r * 32 + hg * 16);
      v8h phi = *(const v8h*)(pst + r * 32 + hg * 16 + 8);
#pragma unroll
      for (int i = 0; i < 8; ++i) { pa[i] = plo[i]; pa[8 + i] = phi[i]; }
    }

    // ---- O += P * V  (V transposed: B[k][n] = Vt[d=n][key=k], contiguous) ----
#pragma unroll
    for (int nt = 0; nt < 8; ++nt) {
      const _Float16* vrow = Vb + (size_t)(nt * 16 + r) * SEQ + k0 + hg * 16;
      v16h bf = *(const v16h*)vrow;
      oacc[nt] = wmma_f16(pa, bf, oacc[nt]);
    }
  }

  // ---- epilogue: normalize by row sum, write fp32 ----
  float* outb = Out + (size_t)b * SEQ * DQK;
#pragma unroll
  for (int nt = 0; nt < 8; ++nt) {
    const int col = nt * 16 + r;
#pragma unroll
    for (int j = 0; j < 8; ++j)
      outb[(size_t)rowq[j] * DQK + col] = oacc[nt][j] / l_i[j];
  }
}

// ---------------------------------------------------------------------------
extern "C" void kernel_launch(void* const* d_in, const int* in_sizes, int n_in,
                              void* d_out, int out_size, void* d_ws, size_t ws_size,
                              hipStream_t stream) {
  (void)in_sizes; (void)n_in; (void)out_size; (void)ws_size;
  const float* q  = (const float*)d_in[0];
  const float* k  = (const float*)d_in[1];
  const float* v  = (const float*)d_in[2];
  const float* Wq = (const float*)d_in[3];
  const float* bq = (const float*)d_in[4];
  const float* Wk = (const float*)d_in[5];
  const float* bk = (const float*)d_in[6];
  const float* Wv = (const float*)d_in[7];
  const float* bv = (const float*)d_in[8];

  _Float16* Qh  = (_Float16*)d_ws;                      // [B*S,128] f16, pre-scaled
  _Float16* Kh  = Qh  + (size_t)BATCH * SEQ * DQK;      // [B*S,128] f16
  _Float16* Vt  = Kh  + (size_t)BATCH * SEQ * DQK;      // [B,128,S] f16 (transposed)
  _Float16* Wqh = Vt  + (size_t)BATCH * SEQ * DQK;      // [128,1024] f16
  _Float16* Wkh = Wqh + (size_t)DQK * DIN;
  _Float16* Wvh = Wkh + (size_t)DQK * DIN;

  const float scale = 1.0f / sqrtf((float)DQK);
  dim3 blk(256);

  const int cvt_blocks = (DQK * DIN) / (256 * 8);       // 64 blocks per matrix
  cvt_f16_kernel<<<dim3(cvt_blocks), blk, 0, stream>>>(Wq, Wqh);
  cvt_f16_kernel<<<dim3(cvt_blocks), blk, 0, stream>>>(Wk, Wkh);
  cvt_f16_kernel<<<dim3(cvt_blocks), blk, 0, stream>>>(Wv, Wvh);

  proj_kernel<<<dim3(128), blk, 0, stream>>>(q, Wqh, bq, Qh, /*transposed=*/0, scale);
  proj_kernel<<<dim3(128), blk, 0, stream>>>(k, Wkh, bk, Kh, /*transposed=*/0, 1.0f);
  proj_kernel<<<dim3(128), blk, 0, stream>>>(v, Wvh, bv, Vt, /*transposed=*/1, 1.0f);
  attn_kernel<<<dim3(BATCH * 16), blk, 0, stream>>>(Qh, Kh, Vt, (float*)d_out);
}
